// TextEncoder_39505109188644
// MI455X (gfx1250) — matrix operand order repaired and checked by reading, not compiled, not run on hardware
//
#include <hip/hip_runtime.h>
#include <math.h>

// ---------------- problem constants ----------------
#define VOCAB_ 30000
#define K_     620
#define KP_    640     // K padded to multiple of 32 for WMMA K-loop
#define D_     2400
#define B_     64
#define T_     128
#define M_     (T_ * B_)   // 8192 rows for every GEMM
#define N0_    (4 * D_)    // 9600, layer 0
#define NL_    (3 * D_)    // 7200, layers 1..3

// ---------------- WMMA types ----------------
typedef __attribute__((ext_vector_type(16))) __bf16 v16bf;
typedef __attribute__((ext_vector_type(8)))  float  v8f;

union Frag { int i[8]; v16bf v; };

// float -> bf16 bits, round-to-nearest-even
__device__ __forceinline__ unsigned short f2bf(float f) {
    union { float f; unsigned int u; } cv; cv.f = f;
    unsigned int u = cv.u;
    u += 0x7FFFu + ((u >> 16) & 1u);
    return (unsigned short)(u >> 16);
}

// ---------------- embedding gather: x[t][b][k] = emb[inp[b][t]][k], bf16, K padded ----------------
__global__ __launch_bounds__(256) void embed_kernel(
    const float* __restrict__ emb, const int* __restrict__ inp,
    unsigned short* __restrict__ X)
{
    int idx = blockIdx.x * 256 + threadIdx.x;              // over M_*KP_
    if (idx >= M_ * KP_) return;
    int row = idx / KP_;
    int kk  = idx - row * KP_;
    float val = 0.f;
    if (kk < K_) {
        int t = row / B_;
        int b = row - t * B_;
        int tok = inp[b * T_ + t];
        val = emb[(size_t)tok * K_ + kk];
    }
    X[idx] = f2bf(val);
}

// ---------------- weight fp32 -> bf16 (layer 0 pads K rows 620->640 with zeros) ----------------
__global__ __launch_bounds__(256) void convert_w0_kernel(
    const float* __restrict__ W, unsigned short* __restrict__ Wb)
{
    int idx = blockIdx.x * 256 + threadIdx.x;              // over KP_*N0_
    if (idx >= KP_ * N0_) return;
    int row = idx / N0_;
    int col = idx - row * N0_;
    float v = (row < K_) ? W[(size_t)row * N0_ + col] : 0.f;
    Wb[idx] = f2bf(v);
}

__global__ __launch_bounds__(256) void convert_w_kernel(
    const float* __restrict__ W, unsigned short* __restrict__ Wb, int total)
{
    int idx = blockIdx.x * 256 + threadIdx.x;
    if (idx >= total) return;
    Wb[idx] = f2bf(W[idx]);
}

// ---------------- bf16 WMMA GEMM: C(MxN) = A(MxKd) * Bm(KdxN), fp32 accumulate ----------------
// block = 256 threads (8 waves). Block tile: 128(M) x 128(N). Wave w owns rows w*16..w*16+15,
// all 128 N columns via 8 accumulators (one A fragment reused 8x per 32-deep K step).
// A tile staged via CDNA5 async global->LDS b128 (ASYNCcnt); B tile via b128 load + LDS transpose.
__global__ __launch_bounds__(256) void gemm_bf16_kernel(
    const unsigned short* __restrict__ A,
    const unsigned short* __restrict__ Bm,
    float* __restrict__ C,
    int Kd, int N)
{
    // 80-byte row stride: 16B-aligned for async b128 stores, conflict-friendly for frag reads
    __shared__ alignas(16) unsigned short As[128][40];   // [m][k]
    __shared__ alignas(16) unsigned short Bs[128][40];   // [n][k] (transposed)

    const int tid  = threadIdx.x;
    const int lane = tid & 31;
    const int wave = tid >> 5;
    const int half = lane >> 4;     // 0: lanes 0-15, 1: lanes 16-31
    const int sub  = lane & 15;     // M index for A-frag, N index for B-frag / C
    const int m0 = blockIdx.y * 128;
    const int n0 = blockIdx.x * 128;

    v8f acc[8];
    #pragma unroll
    for (int j = 0; j < 8; ++j) acc[j] = (v8f){0.f,0.f,0.f,0.f,0.f,0.f,0.f,0.f};

    for (int k0 = 0; k0 < Kd; k0 += 32) {
        // ---- A tile: 128x32 bf16 = 512 x 16B chunks, 2 per thread, async into LDS ----
        #pragma unroll
        for (int i = 0; i < 2; ++i) {
            int chunk = tid + i * 256;
            int row = chunk >> 2;           // m within tile
            int c16 = (chunk & 3) << 3;     // k element offset (0,8,16,24)
            unsigned long long ga =
                (unsigned long long)(const void*)(A + (size_t)(m0 + row) * Kd + (k0 + c16));
            unsigned la = (unsigned)(size_t)(void*)&As[row][c16];
            asm volatile("global_load_async_to_lds_b128 %0, %1, off"
                         :: "v"(la), "v"(ga) : "memory");
        }
        // ---- B tile: 32(k) x 128(n) = 512 x 16B chunks, 2 per thread; transpose into Bs[n][k] ----
        #pragma unroll
        for (int i = 0; i < 2; ++i) {
            int chunk = tid + i * 256;
            int kr = chunk >> 4;            // k row within tile
            int nc = (chunk & 15) << 3;     // n element offset (0..120)
            if (n0 + nc + 8 <= N) {         // N multiple of 8 -> chunk all-or-nothing
                int4 w = *(const int4*)(Bm + (size_t)(k0 + kr) * N + (n0 + nc));
                Bs[nc + 0][kr] = (unsigned short)(w.x);
                Bs[nc + 1][kr] = (unsigned short)(((unsigned)w.x) >> 16);
                Bs[nc + 2][kr] = (unsigned short)(w.y);
                Bs[nc + 3][kr] = (unsigned short)(((unsigned)w.y) >> 16);
                Bs[nc + 4][kr] = (unsigned short)(w.z);
                Bs[nc + 5][kr] = (unsigned short)(((unsigned)w.z) >> 16);
                Bs[nc + 6][kr] = (unsigned short)(w.w);
                Bs[nc + 7][kr] = (unsigned short)(((unsigned)w.w) >> 16);
            } else {
                #pragma unroll
                for (int q = 0; q < 8; ++q) Bs[nc + q][kr] = 0;
            }
        }
        asm volatile("s_wait_asynccnt 0x0" ::: "memory");   // our async A writes landed
        __syncthreads();

        // A fragment per ISA 16-bit A 16x32 layout:
        // lanes 0-15: v0..3 -> K 0..7, v4..7 -> K 16..23; lanes 16-31: K 8..15 / 24..31
        Frag fa;
        const int mrow = wave * 16 + sub;
        #pragma unroll
        for (int r = 0; r < 8; ++r) {
            int kp = ((r & 3) << 1) + ((r >> 2) << 4) + (half << 3);
            fa.i[r] = *(const int*)&As[mrow][kp];
        }
        // 8 x WMMA across the 128-wide N strip, A fragment reused
        #pragma unroll
        for (int j = 0; j < 8; ++j) {
            Frag fb;
            const int ncol = j * 16 + sub;
            #pragma unroll
            for (int r = 0; r < 8; ++r)
                fb.i[r] = *(const int*)&Bs[ncol][((r & 3) << 1) + ((r >> 2) << 4) + (half << 3)];
            acc[j] = __builtin_amdgcn_wmma_f32_16x16x32_bf16(
                        false, fa.v, false, fb.v, (short)0, acc[j], false, false);
        }
        __syncthreads();
    }

    // store: C/D layout — lane sub = N, vgpr r -> M = half*8 + r
    #pragma unroll
    for (int j = 0; j < 8; ++j) {
        int n = n0 + j * 16 + sub;
        if (n < N) {
            #pragma unroll
            for (int r = 0; r < 8; ++r) {
                int m = m0 + wave * 16 + (half << 3) + r;
                C[(size_t)m * N + n] = acc[j][r];
            }
        }
    }
}

// ---------------- SRU recurrence scan over T (sequential), parallel over B*D ----------------
// U: (T, B, kcols) fp32. Xp == nullptr => k=4 (x' = U[...,3D:4D]) else x' = Xp (T,B,D).
__global__ __launch_bounds__(256) void sru_scan_kernel(
    const float* __restrict__ U, const float* __restrict__ bias,
    const float* __restrict__ Xp,
    float* __restrict__ H32, unsigned short* __restrict__ H16,
    int kcols)
{
    int idx = blockIdx.x * 256 + threadIdx.x;
    if (idx >= B_ * D_) return;
    int b = idx / D_;
    int d = idx - b * D_;
    const float bf_ = bias[d];
    const float br_ = bias[D_ + d];
    float c = 0.f;
    for (int t = 0; t < T_; ++t) {
        size_t rb = ((size_t)t * B_ + b) * kcols;
        float xt = U[rb + d];
        float f  = 1.f / (1.f + __expf(-(U[rb + D_ + d] + bf_)));
        float r  = 1.f / (1.f + __expf(-(U[rb + 2 * D_ + d] + br_)));
        size_t ho = ((size_t)t * B_ + b) * D_ + d;
        float xp = Xp ? Xp[ho] : U[rb + 3 * D_ + d];
        c = f * c + (1.f - f) * xt;
        float h = r * tanhf(c) + (1.f - r) * xp;
        H32[ho] = h;
        H16[ho] = f2bf(h);
    }
}

// ---------------- final row L2-normalize of h[T-1] ----------------
__global__ __launch_bounds__(256) void norm_kernel(
    const float* __restrict__ H32, float* __restrict__ out)
{
    __shared__ float red[256];
    int b = blockIdx.x;
    const float* row = H32 + ((size_t)(T_ - 1) * B_ + b) * D_;
    float s = 0.f;
    for (int d = threadIdx.x; d < D_; d += 256) { float v = row[d]; s += v * v; }
    red[threadIdx.x] = s;
    __syncthreads();
    for (int off = 128; off > 0; off >>= 1) {
        if (threadIdx.x < off) red[threadIdx.x] += red[threadIdx.x + off];
        __syncthreads();
    }
    float inv = rsqrtf(red[0]);
    for (int d = threadIdx.x; d < D_; d += 256) out[(size_t)b * D_ + d] = row[d] * inv;
}

// ---------------- launcher ----------------
extern "C" void kernel_launch(void* const* d_in, const int* in_sizes, int n_in,
                              void* d_out, int out_size, void* d_ws, size_t ws_size,
                              hipStream_t stream) {
    (void)in_sizes; (void)n_in; (void)out_size; (void)ws_size;
    const float* emb = (const float*)d_in[0];
    const float* Wl[4] = { (const float*)d_in[1], (const float*)d_in[3],
                           (const float*)d_in[5], (const float*)d_in[7] };
    const float* bl[4] = { (const float*)d_in[2], (const float*)d_in[4],
                           (const float*)d_in[6], (const float*)d_in[8] };
    const int* inp = (const int*)d_in[9];
    float* out = (float*)d_out;

    // workspace carve-up (256B aligned)
    char* ws = (char*)d_ws;
    size_t off = 0;
    auto carve = [&](size_t bytes) { void* p = ws + off; off += (bytes + 255) & ~(size_t)255; return p; };
    float*          U    = (float*)         carve((size_t)M_ * N0_ * sizeof(float));      // 314.6 MB
    float*          H32a = (float*)         carve((size_t)M_ * D_  * sizeof(float));      //  78.6 MB
    float*          H32b = (float*)         carve((size_t)M_ * D_  * sizeof(float));      //  78.6 MB
    unsigned short* H16  = (unsigned short*)carve((size_t)M_ * D_  * sizeof(short));      //  39.3 MB
    unsigned short* X0b  = (unsigned short*)carve((size_t)M_ * KP_ * sizeof(short));      //  10.5 MB
    unsigned short* W0b  = (unsigned short*)carve((size_t)KP_ * N0_ * sizeof(short));     //  12.3 MB
    unsigned short* Wb   = (unsigned short*)carve((size_t)D_  * NL_ * sizeof(short));     //  34.6 MB

    // 1) embedding gather -> bf16 A matrix (M x KP_)
    embed_kernel<<<(M_ * KP_ + 255) / 256, 256, 0, stream>>>(emb, inp, X0b);

    // 2) layer 0: W convert, GEMM (Kd=640, N=9600), scan (k=4)
    convert_w0_kernel<<<(KP_ * N0_ + 255) / 256, 256, 0, stream>>>(Wl[0], W0b);
    {
        dim3 grid(N0_ / 128, M_ / 128);   // 75 x 64
        gemm_bf16_kernel<<<grid, 256, 0, stream>>>(X0b, W0b, U, KP_, N0_);
    }
    sru_scan_kernel<<<(B_ * D_ + 255) / 256, 256, 0, stream>>>(U, bl[0], nullptr, H32a, H16, N0_);

    // 3) layers 1..3: Kd=2400, N=7200, k=3 (x' = layer input, fp32)
    float* Hin = H32a;
    float* Hout = H32b;
    for (int l = 1; l <= 3; ++l) {
        convert_w_kernel<<<(D_ * NL_ + 255) / 256, 256, 0, stream>>>(Wl[l], Wb, D_ * NL_);
        dim3 grid((NL_ + 127) / 128, M_ / 128);   // 57 x 64
        gemm_bf16_kernel<<<grid, 256, 0, stream>>>(H16, Wb, U, D_, NL_);
        sru_scan_kernel<<<(B_ * D_ + 255) / 256, 256, 0, stream>>>(U, bl[l], Hin, Hout, H16, NL_);
        float* tmp = Hin; Hin = Hout; Hout = tmp;   // Hin now holds this layer's output
    }

    // 4) normalize last timestep (layer 3 output ended up in Hin after the swap)
    norm_kernel<<<B_, 256, 0, stream>>>(Hin, out);
}
